// BasePointProcess_12687333392939
// MI455X (gfx1250) — compile-verified, matrix operand order checked by reading, not run on hardware
//
#include <hip/hip_runtime.h>
#include <hip/hip_bf16.h>
#include <math.h>

typedef __attribute__((ext_vector_type(16))) __bf16 v16bf;
typedef __attribute__((ext_vector_type(8)))  __bf16 v8bf;
typedef __attribute__((ext_vector_type(8)))  float  v8f;

#define SEQ   256
#define SGRID 64
#define NT    64
#define BATCH 32

__device__ __forceinline__ float softplus_f(float x) {
    // log(1 + exp(x)) = max(x,0) + log1p(exp(-|x|))
    float ax = fabsf(x);
    return fmaxf(x, 0.0f) + log1pf(__expf(-ax));
}

// ---------------------------------------------------------------------------
// Kernel 1: lams[b,i] = softplus(sum_{j<i, t_j>0} alpha*exp(-beta*dt - gamma*d2) + mu) + 1e-5
// Also writes sumlog[b] = sum_i log(lams[b,i]) * (t_i > 0) into workspace.
// One block per batch, 256 threads (thread i = event i).
// ---------------------------------------------------------------------------
__global__ __launch_bounds__(256) void pp_lams_kernel(
    const float* __restrict__ X,
    const float* __restrict__ mu_p, const float* __restrict__ alpha_p,
    const float* __restrict__ beta_p, const float* __restrict__ gamma_p,
    float* __restrict__ out, float* __restrict__ sumlog_ws)
{
    __shared__ float tS[SEQ], sxS[SEQ], syS[SEQ];
    __shared__ float red[8];

    const int b = blockIdx.x;
    const int i = threadIdx.x;
    const float mu = *mu_p, alpha = *alpha_p, beta = *beta_p, gamma = *gamma_p;
    const float* Xb = X + b * SEQ * 3;

    tS[i]  = Xb[i * 3 + 0];
    sxS[i] = Xb[i * 3 + 1];
    syS[i] = Xb[i * 3 + 2];
    __syncthreads();

    const float ti = tS[i], sxi = sxS[i], syi = syS[i];
    float acc = 0.0f;
    for (int j = 0; j < i; ++j) {
        float tj = tS[j];
        if (tj > 0.0f) {
            float dt = ti - tj;
            float dx = sxi - sxS[j];
            float dy = syi - syS[j];
            acc += __expf(-beta * dt - gamma * (dx * dx + dy * dy));
        }
    }
    acc *= alpha;

    float lam = softplus_f(acc + mu) + 1e-5f;
    out[b * SEQ + i] = lam;

    float lg = (ti > 0.0f) ? __logf(lam) : 0.0f;
    #pragma unroll
    for (int off = 16; off > 0; off >>= 1)
        lg += __shfl_down(lg, off, 32);
    const int lane = i & 31, wv = i >> 5;
    if (lane == 0) red[wv] = lg;
    __syncthreads();
    if (i == 0) {
        float s = 0.0f;
        #pragma unroll
        for (int w = 0; w < 8; ++w) s += red[w];
        sumlog_ws[b] = s;
    }
}

// ---------------------------------------------------------------------------
// Kernel 2: grid integral via WMMA (bf16 -> f32).
//   Per (batch, timestep):  C[64x64] = (E1 .* tk)[64x256] x E2^T[256x64]
//   gridsum[b] = sum_{t,g} softplus(C + mu);  loglik[b] = sumlog[b] - gridsum*unit_vol
// One block per batch, 512 threads = 16 waves = 4x4 grid of 16x16 output tiles.
// Both E1 (row tile) and E2 (col tile) fragments are time-invariant and live
// entirely in registers; per-timestep work is 2 ds_load_b128 + 8 v_pk_mul_bf16
// per k-step feeding v_wmma_f32_16x16x32_bf16.
// ---------------------------------------------------------------------------
__global__ __launch_bounds__(512) void pp_grid_wmma_kernel(
    const float* __restrict__ X,
    const float* __restrict__ mu_p, const float* __restrict__ alpha_p,
    const float* __restrict__ beta_p, const float* __restrict__ gamma_p,
    float* __restrict__ out, const float* __restrict__ sumlog_ws)
{
    __shared__ float tS[SEQ], sxS[SEQ], syS[SEQ];
    __shared__ __align__(16) __bf16 tkB[SEQ];   // temporal kernel, packed bf16
    __shared__ float red[16];

    const int b   = blockIdx.x;
    const int tid = threadIdx.x;
    const float mu = *mu_p, alpha = *alpha_p, beta = *beta_p, gamma = *gamma_p;
    const float* Xb = X + b * SEQ * 3;

    if (tid < SEQ) {
        tS[tid]  = Xb[tid * 3 + 0];
        sxS[tid] = Xb[tid * 3 + 1];
        syS[tid] = Xb[tid * 3 + 2];
    }
    __syncthreads();

    const int wv     = tid >> 5;
    const int lane   = tid & 31;
    const int half   = lane >> 4;       // K half select per ISA 16-bit layout
    const int lane16 = lane & 15;
    const int ti = wv >> 2;             // output row tile 0..3
    const int tj = wv & 3;              // output col tile 0..3

    // Time-invariant register fragments.
    // 16-bit fragment element e -> K = 16*(e>>3) + 8*half + (e&7)  (ISA 7.12.2).
    //   A (E1): M = lane16, row = ti*16+lane16, spatial-x kernel
    //   B (E2): N = lane16, col = tj*16+lane16, spatial-y kernel
    v16bf bfrag[8];   // E2 tile, full K=256 in 8 k-steps
    v16bf e1c[8];     // E1 row slices in A-fragment element order
    {
        const float ssi = (float)(ti * 16 + lane16) * (1.0f / 63.0f);
        const float ssj = (float)(tj * 16 + lane16) * (1.0f / 63.0f);
        #pragma unroll
        for (int k = 0; k < 8; ++k) {
            #pragma unroll
            for (int e = 0; e < 16; ++e) {
                const int K = 16 * (e >> 3) + 8 * half + (e & 7);
                const int s = k * 32 + K;
                float da = ssi - sxS[s];
                float db = ssj - syS[s];
                e1c[k][e]   = (__bf16)__expf(-gamma * da * da);
                bfrag[k][e] = (__bf16)__expf(-gamma * db * db);
            }
        }
    }

    float gsum = 0.0f;

    for (int step = 0; step < NT; ++step) {
        const float tc = (float)step * (1.0f / 63.0f);
        if (tid < SEQ) {
            float tv = tS[tid];
            float tk = (tv > 0.0f && tv <= tc) ? alpha * __expf(-beta * (tc - tv)) : 0.0f;
            tkB[tid] = (__bf16)tk;
        }
        __syncthreads();

        v8f c = {};
        #pragma unroll
        for (int k = 0; k < 8; ++k) {
            const int s0 = k * 32 + 8 * half;
            union { v16bf v; unsigned int u[8]; } a, e1;
            union { v8bf  v; unsigned int u[4]; } tlo, thi;
            e1.v  = e1c[k];
            tlo.v = *(const v8bf*)&tkB[s0];        // tk[s0 .. s0+7]
            thi.v = *(const v8bf*)&tkB[s0 + 16];   // tk[s0+16 .. s0+23]
            #pragma unroll
            for (int d = 0; d < 4; ++d) {
                asm("v_pk_mul_bf16 %0, %1, %2"
                    : "=v"(a.u[d]) : "v"(e1.u[d]), "v"(tlo.u[d]));
                asm("v_pk_mul_bf16 %0, %1, %2"
                    : "=v"(a.u[d + 4]) : "v"(e1.u[d + 4]), "v"(thi.u[d]));
            }
            c = __builtin_amdgcn_wmma_f32_16x16x32_bf16(
                    false, a.v, false, bfrag[k], (short)0, c, false, false);
        }

        #pragma unroll
        for (int e = 0; e < 8; ++e)
            gsum += softplus_f(c[e] + mu);
        __syncthreads();   // protect tkB before next timestep overwrite
    }

    // Reduce gsum over 512 threads.
    #pragma unroll
    for (int off = 16; off > 0; off >>= 1)
        gsum += __shfl_down(gsum, off, 32);
    if (lane == 0) red[wv] = gsum;
    __syncthreads();
    if (tid == 0) {
        float tot = 0.0f;
        #pragma unroll
        for (int w = 0; w < 16; ++w) tot += red[w];
        const float unit_vol = 1.0f / (64.0f * 64.0f * 64.0f);
        out[BATCH * SEQ + b] = sumlog_ws[b] - tot * unit_vol;
    }
}

extern "C" void kernel_launch(void* const* d_in, const int* in_sizes, int n_in,
                              void* d_out, int out_size, void* d_ws, size_t ws_size,
                              hipStream_t stream) {
    const float* X     = (const float*)d_in[0];
    const float* mu    = (const float*)d_in[1];
    const float* alpha = (const float*)d_in[2];
    const float* beta  = (const float*)d_in[3];
    const float* gamma = (const float*)d_in[4];
    float* out    = (float*)d_out;
    float* sumlog = (float*)d_ws;   // 32 floats of scratch

    pp_lams_kernel<<<BATCH, 256, 0, stream>>>(X, mu, alpha, beta, gamma, out, sumlog);
    pp_grid_wmma_kernel<<<BATCH, 512, 0, stream>>>(X, mu, alpha, beta, gamma, out, sumlog);
}
